// EGTLayer_40140764348626
// MI455X (gfx1250) — matrix operand order repaired and checked by reading, not compiled, not run on hardware
//
#include <hip/hip_runtime.h>
#include <hip/hip_bf16.h>

typedef _Float16 v16h __attribute__((ext_vector_type(16)));
typedef _Float16 v8h  __attribute__((ext_vector_type(8)));
typedef float    v8f  __attribute__((ext_vector_type(8)));

// ---------------------------------------------------------------------------
// WMMA fragment helpers for V_WMMA_F32_16X16X32_F16 (wave32).
// A (16x32, row-major in memory): lane L holds row (L&15); half j is
// K = 16*(L>>4)+j  -> two b128 loads.
// B (32x16): lane L holds col (L&15); half j is row K = 16*(L>>4)+j.
// We PRE-PACK B into per-lane 32B chunks so the load is one contiguous b128x2.
// C/D f32: VGPR r = row r + 8*(L>>4), col = L&15.
// ---------------------------------------------------------------------------
__device__ __forceinline__ v16h load_a_frag(const _Float16* A, int lda, int row0, int k0) {
    const int lane = threadIdx.x & 31;
    const _Float16* p = A + (long long)(row0 + (lane & 15)) * lda + k0 + ((lane >> 4) << 4);
    v8h lo = *(const v8h*)p;
    v8h hi = *(const v8h*)(p + 8);
    v16h r;
#pragma unroll
    for (int i = 0; i < 8; ++i) { r[i] = lo[i]; r[8 + i] = hi[i]; }
    return r;
}

__device__ __forceinline__ v16h load_b_packed(const _Float16* tileBase) {
    const int lane = threadIdx.x & 31;
    return *(const v16h*)(tileBase + lane * 16);
}

__device__ __forceinline__ float eluf(float v) { return v > 0.f ? v : __expf(v) - 1.f; }

__device__ __forceinline__ float wave_max(float v) {
#pragma unroll
    for (int o = 16; o > 0; o >>= 1) v = fmaxf(v, __shfl_xor(v, o, 32));
    return v;
}
__device__ __forceinline__ float wave_sum(float v) {
#pragma unroll
    for (int o = 16; o > 0; o >>= 1) v += __shfl_xor(v, o, 32);
    return v;
}

// ---------------------------------------------------------------------------
// Pack a row-major f32 weight K x N into per-lane B-fragment order (f16).
// Fragment tile t = kt*(N/16)+nt occupies halves [t*512, t*512+512).
// ---------------------------------------------------------------------------
__global__ void pack_b(const float* __restrict__ W, _Float16* __restrict__ Wp, int K, int N) {
    int idx = blockIdx.x * 256 + threadIdx.x;
    if (idx >= K * N) return;
    int j = idx & 15;
    int lane = (idx >> 4) & 31;
    int t = idx >> 9;
    int ntiles = N >> 4;
    int nt = t % ntiles, kt = t / ntiles;
    int kk = kt * 32 + ((lane >> 4) << 4) + j;
    int nn = nt * 16 + (lane & 15);
    Wp[idx] = (_Float16)W[(long long)kk * N + nn];
}

// ---------------------------------------------------------------------------
// Row LayerNorm (f32 in, f16 out), one block per row.
// ---------------------------------------------------------------------------
__global__ void ln_rows(const float* __restrict__ X, const float* __restrict__ g,
                        const float* __restrict__ bb, _Float16* __restrict__ out, int cols) {
    __shared__ float s1[256], s2[256];
    const float* x = X + (size_t)blockIdx.x * cols;
    float s = 0.f, q = 0.f;
    for (int c = threadIdx.x; c < cols; c += 256) { float v = x[c]; s += v; q += v * v; }
    s1[threadIdx.x] = s; s2[threadIdx.x] = q;
    __syncthreads();
    for (int off = 128; off > 0; off >>= 1) {
        if (threadIdx.x < off) { s1[threadIdx.x] += s1[threadIdx.x + off]; s2[threadIdx.x] += s2[threadIdx.x + off]; }
        __syncthreads();
    }
    float mu = s1[0] / cols;
    float var = s2[0] / cols - mu * mu;
    float rs = rsqrtf(var + 1e-5f);
    _Float16* o = out + (size_t)blockIdx.x * cols;
    for (int c = threadIdx.x; c < cols; c += 256)
        o[c] = (_Float16)((x[c] - mu) * rs * g[c] + bb[c]);
}

// ---------------------------------------------------------------------------
// Generic WMMA GEMM: out = epi(A_f16[MxK] @ Bpacked[KxN] + bias [+ res]).
// 8 waves/block, one 16x16 output tile per wave.
// ---------------------------------------------------------------------------
template <bool RES, bool DO_ELU, bool W32, bool W16>
__global__ void gemm_wmma(const _Float16* __restrict__ A, const _Float16* __restrict__ Bp,
                          const float* __restrict__ bias, const float* __restrict__ res,
                          float* __restrict__ out32, _Float16* __restrict__ out16,
                          int M, int N, int K) {
    const int w = threadIdx.x >> 5, lane = threadIdx.x & 31;
    const int tile = blockIdx.x * 8 + w;
    const int ntiles = N >> 4;
    const int mt = tile / ntiles, nt = tile - mt * ntiles;
    const int ktiles = K >> 5;
    v8f acc = {};
    for (int kt = 0; kt < ktiles; ++kt) {
        v16h a = load_a_frag(A, K, mt * 16, kt * 32);
        v16h b = load_b_packed(Bp + ((long long)(kt * ntiles + nt) << 9));
        acc = __builtin_amdgcn_wmma_f32_16x16x32_f16(false, a, false, b, (short)0, acc, false, false);
    }
    const int col = nt * 16 + (lane & 15);
    const int rbase = mt * 16 + ((lane >> 4) << 3);
    const float bv = bias[col];
#pragma unroll
    for (int r = 0; r < 8; ++r) {
        long long o = (long long)(rbase + r) * N + col;
        float v = acc[r] + bv;
        if (RES) v += res[o];
        if (DO_ELU) v = eluf(v);
        if (W32) out32[o] = v;
        if (W16) out16[o] = (_Float16)v;
    }
}

// ---------------------------------------------------------------------------
// Split QKV (f32, col = part*1024 + d*16 + h) into per-(b,h) f16 tensors:
//   Qf [bh][l*64+d]  (A-side row-major)
//   KTp[bh]          (K^T 64x512 in packed B-frag order, 32 m-tiles)
//   Vp [bh]          (V 512x64 in packed B-frag order, 4 n-tiles)
// ---------------------------------------------------------------------------
__global__ void prep_qkv(const float* __restrict__ QKV, _Float16* __restrict__ Qf,
                         _Float16* __restrict__ KTp, _Float16* __restrict__ Vp) {
    const int bh = blockIdx.y;
    const int b = bh >> 4, h = bh & 15;
    const float* Q = QKV + (long long)b * 512 * 3072;
    int idx = blockIdx.x * 256 + threadIdx.x;        // 0 .. 3*32768
    int sec = idx >> 15, t = idx & 32767;
    long long base = (long long)bh * 32768;
    if (sec == 0) {
        int l = t >> 6, d = t & 63;
        Qf[base + t] = (_Float16)Q[(long long)l * 3072 + d * 16 + h];
    } else if (sec == 1) {
        int j = t & 15, lane = (t >> 4) & 31, tile = t >> 9;
        int mt = tile & 31, kt = tile >> 5;
        int d = kt * 32 + ((lane >> 4) << 4) + j;
        int m = mt * 16 + (lane & 15);
        KTp[base + t] = (_Float16)Q[(long long)m * 3072 + 1024 + d * 16 + h];
    } else {
        int j = t & 15, lane = (t >> 4) & 31, tile = t >> 9;
        int nt = tile & 3, mt = tile >> 2;
        int m = mt * 32 + ((lane >> 4) << 4) + j;
        int k = nt * 16 + (lane & 15);
        Vp[base + t] = (_Float16)Q[(long long)m * 3072 + 2048 + k * 16 + h];
    }
}

// ---------------------------------------------------------------------------
// e -> LN -> E = e_ln@W_E + b_E, gate = sigmoid(e_ln@W_G + b_G).
// Writes scattered into [b,h,l,m] layout so all consumers stream coalesced.
// Block: 256 threads, 64 (b,l,m) rows staged through LDS.
// ---------------------------------------------------------------------------
__global__ void eg_kernel(const float* __restrict__ e, const float* __restrict__ g_ln,
                          const float* __restrict__ b_ln, const float* __restrict__ W_E,
                          const float* __restrict__ b_E, const float* __restrict__ W_G,
                          const float* __restrict__ b_G, float* __restrict__ ET,
                          float* __restrict__ GT) {
    __shared__ float es[64 * 64];
    __shared__ float we[64 * 16], wg[64 * 16];
    const long long row0 = (long long)blockIdx.x * 64;
    const float* src = e + row0 * 64;
    for (int i = threadIdx.x; i < 4096; i += 256) es[i] = src[i];
    for (int i = threadIdx.x; i < 1024; i += 256) { we[i] = W_E[i]; wg[i] = W_G[i]; }
    __syncthreads();
    const int r = threadIdx.x >> 2, sub = threadIdx.x & 3;
    float s = 0.f, q = 0.f;
    for (int d = 0; d < 64; ++d) { float v = es[r * 64 + d]; s += v; q += v * v; }
    float mu = s * (1.f / 64), var = q * (1.f / 64) - mu * mu, rs = rsqrtf(var + 1e-5f);
    long long R = row0 + r;
    int b = (int)(R >> 18), l = (int)((R >> 9) & 511), m = (int)(R & 511);
    for (int hh = sub * 4; hh < sub * 4 + 4; ++hh) {
        float aE = 0.f, aG = 0.f;
        for (int d = 0; d < 64; ++d) {
            float xe = (es[r * 64 + d] - mu) * rs * g_ln[d] + b_ln[d];
            aE += xe * we[d * 16 + hh];
            aG += xe * wg[d * 16 + hh];
        }
        aE += b_E[hh]; aG += b_G[hh];
        long long o = ((((long long)b * 16 + hh) * 512 + l) << 9) + m;
        ET[o] = aE;
        GT[o] = 1.f / (1.f + __expf(-aG));
    }
}

// ---------------------------------------------------------------------------
// H_hat[b,h,l,m] = clip(Q@K^T / 8, -5, 5) + E.  One 16x16 tile per wave.
// ---------------------------------------------------------------------------
__global__ void scores_kernel(const _Float16* __restrict__ Qf, const _Float16* __restrict__ KTp,
                              const float* __restrict__ ET, float* __restrict__ Hh) {
    const int bh = blockIdx.y;
    const int w = threadIdx.x >> 5, lane = threadIdx.x & 31;
    const int tile = blockIdx.x * 8 + w;          // 1024 tiles: lt(32) x mt(32)
    const int lt = tile >> 5, mt = tile & 31;
    const _Float16* Qb = Qf + (long long)bh * 32768;
    const _Float16* Kb = KTp + (long long)bh * 32768;
    v8f acc = {};
#pragma unroll
    for (int kt = 0; kt < 2; ++kt) {
        v16h a = load_a_frag(Qb, 64, lt * 16, kt * 32);
        v16h b = load_b_packed(Kb + ((long long)(kt * 32 + mt) << 9));
        acc = __builtin_amdgcn_wmma_f32_16x16x32_f16(false, a, false, b, (short)0, acc, false, false);
    }
    const long long pbase = (long long)bh * 262144;
    const int mcol = mt * 16 + (lane & 15);
    const int rbase = lt * 16 + ((lane >> 4) << 3);
#pragma unroll
    for (int r = 0; r < 8; ++r) {
        long long o = pbase + (long long)(rbase + r) * 512 + mcol;
        float s = acc[r] * 0.125f;                // DOT_DIM^-0.5
        s = fminf(fmaxf(s, -5.f), 5.f);
        Hh[o] = s + ET[o];
    }
}

// ---------------------------------------------------------------------------
// A_tild = softmax_m(H_hat) * gate.  One wave per (b,h,l) row of 512.
// ---------------------------------------------------------------------------
__global__ void softmax_kernel(const float* __restrict__ Hh, const float* __restrict__ GT,
                               _Float16* __restrict__ At) {
    const int w = threadIdx.x >> 5, lane = threadIdx.x & 31;
    const long long row = (long long)blockIdx.x * 8 + w;   // bh*512 + l
    const float* x = Hh + row * 512;
    float vals[16];
    float mx = -1e30f;
#pragma unroll
    for (int i = 0; i < 16; ++i) { vals[i] = x[lane + 32 * i]; mx = fmaxf(mx, vals[i]); }
    mx = wave_max(mx);
    float sm = 0.f;
#pragma unroll
    for (int i = 0; i < 16; ++i) { vals[i] = __expf(vals[i] - mx); sm += vals[i]; }
    sm = wave_sum(sm);
    const float inv = 1.f / sm;
    const float* gt = GT + row * 512;
    _Float16* o = At + row * 512;
#pragma unroll
    for (int i = 0; i < 16; ++i) o[lane + 32 * i] = (_Float16)(vals[i] * inv * gt[lane + 32 * i]);
}

// ---------------------------------------------------------------------------
// V_att = A_tild @ V per (b,h); write f16 straight into [b,l, k*16+h] layout
// (the reference's reshape(B,N,NDIM)).
// ---------------------------------------------------------------------------
__global__ void attnv_kernel(const _Float16* __restrict__ At, const _Float16* __restrict__ Vp,
                             _Float16* __restrict__ Vatt) {
    const int bh = blockIdx.y, b = bh >> 4, h = bh & 15;
    const int w = threadIdx.x >> 5, lane = threadIdx.x & 31;
    const int tile = blockIdx.x * 8 + w;          // 128 tiles: lt(32) x nt(4)
    const int lt = tile >> 2, nt = tile & 3;
    const _Float16* Ab = At + (long long)bh * 262144;
    const _Float16* Vb = Vp + (long long)bh * 32768;
    v8f acc = {};
    for (int kt = 0; kt < 16; ++kt) {
        v16h a = load_a_frag(Ab, 512, lt * 16, kt * 32);
        v16h bb = load_b_packed(Vb + ((long long)(kt * 4 + nt) << 9));
        acc = __builtin_amdgcn_wmma_f32_16x16x32_f16(false, a, false, bb, (short)0, acc, false, false);
    }
    const int kcol = nt * 16 + (lane & 15);
    const int rbase = lt * 16 + ((lane >> 4) << 3);
#pragma unroll
    for (int r = 0; r < 8; ++r) {
        int l = rbase + r;
        Vatt[((long long)(b * 512 + l)) * 1024 + kcol * 16 + h] = (_Float16)acc[r];
    }
}

// ---------------------------------------------------------------------------
// e_out1[b,l,m,k] = sum_h H_hat[b,h,l,m]*W_Oe[h,k] + b_Oe[k] + e[...]
// (K=16 < 32, so plain VALU; purely bandwidth bound anyway.)
// ---------------------------------------------------------------------------
__global__ void eout1_kernel(const float* __restrict__ Hh, const float* __restrict__ e,
                             const float* __restrict__ W_Oe, const float* __restrict__ b_Oe,
                             float* __restrict__ out) {
    __shared__ float woe[1024];
    __shared__ float hs[4][16];
    for (int i = threadIdx.x; i < 1024; i += 256) woe[i] = W_Oe[i];
    const long long blk = blockIdx.x;             // (b*512 + l)*128 + mb
    const int mb = (int)(blk & 127);
    const int l = (int)((blk >> 7) & 511);
    const int b = (int)(blk >> 16);
    const int mloc = threadIdx.x >> 6, k = threadIdx.x & 63;
    const int m = mb * 4 + mloc;
    if (k < 16) hs[mloc][k] = Hh[((((long long)b * 16 + k) * 512 + l) << 9) + m];
    __syncthreads();
    float acc = b_Oe[k];
#pragma unroll
    for (int hh = 0; hh < 16; ++hh) acc += hs[mloc][hh] * woe[hh * 64 + k];
    long long o = ((((long long)b * 512 + l) << 9) + m) * 64 + k;
    out[o] = acc + e[o];
}

// ---------------------------------------------------------------------------
// Fused e-FFN, in place on d_out: x = LN(eo); eo += elu(x@W_e1+b1)@W_e2+b2.
// One 16-row tile per block (128 threads = 4 waves), WMMA for both GEMMs.
// ---------------------------------------------------------------------------
__global__ void effn_kernel(float* __restrict__ eo, const float* __restrict__ g,
                            const float* __restrict__ bln, const _Float16* __restrict__ We1p,
                            const float* __restrict__ b_e1, const _Float16* __restrict__ We2p,
                            const float* __restrict__ b_e2) {
    __shared__ __align__(16) float stage[16 * 64];
    __shared__ __align__(32) _Float16 af[16 * 64];
    __shared__ __align__(32) _Float16 hbuf[16 * 128];
    const long long row0 = (long long)blockIdx.x * 16;
    float* base = eo + row0 * 64;
    for (int i = threadIdx.x; i < 1024; i += 128) stage[i] = base[i];
    __syncthreads();
    if (threadIdx.x < 16) {
        const int r = threadIdx.x;
        float s = 0.f, q = 0.f;
        for (int d = 0; d < 64; ++d) { float v = stage[r * 64 + d]; s += v; q += v * v; }
        float mu = s * (1.f / 64), var = q * (1.f / 64) - mu * mu, rs = rsqrtf(var + 1e-5f);
        for (int d = 0; d < 64; ++d)
            af[r * 64 + d] = (_Float16)((stage[r * 64 + d] - mu) * rs * g[d] + bln[d]);
    }
    __syncthreads();
    const int w = threadIdx.x >> 5, lane = threadIdx.x & 31;
    // GEMM1 + ELU: 16x128, 2 tiles per wave
    for (int tt = 0; tt < 2; ++tt) {
        const int nt = w * 2 + tt;
        v8f acc = {};
#pragma unroll
        for (int kt = 0; kt < 2; ++kt) {
            v16h a = load_a_frag(af, 64, 0, kt * 32);
            v16h b = load_b_packed(We1p + ((long long)(kt * 8 + nt) << 9));
            acc = __builtin_amdgcn_wmma_f32_16x16x32_f16(false, a, false, b, (short)0, acc, false, false);
        }
        const int col = nt * 16 + (lane & 15);
        const int rb = (lane >> 4) << 3;
#pragma unroll
        for (int r = 0; r < 8; ++r)
            hbuf[(rb + r) * 128 + col] = (_Float16)eluf(acc[r] + b_e1[col]);
    }
    __syncthreads();
    // GEMM2 + bias + residual: 16x64, 1 tile per wave
    {
        const int nt = w;
        v8f acc = {};
#pragma unroll
        for (int kt = 0; kt < 4; ++kt) {
            v16h a = load_a_frag(hbuf, 128, 0, kt * 32);
            v16h b = load_b_packed(We2p + ((long long)(kt * 4 + nt) << 9));
            acc = __builtin_amdgcn_wmma_f32_16x16x32_f16(false, a, false, b, (short)0, acc, false, false);
        }
        const int col = nt * 16 + (lane & 15);
        const int rb = (lane >> 4) << 3;
#pragma unroll
        for (int r = 0; r < 8; ++r) {
            int rr = rb + r;
            base[rr * 64 + col] = acc[r] + b_e2[col] + stage[rr * 64 + col];
        }
    }
}

// ---------------------------------------------------------------------------
extern "C" void kernel_launch(void* const* d_in, const int* in_sizes, int n_in,
                              void* d_out, int out_size, void* d_ws, size_t ws_size,
                              hipStream_t stream) {
    (void)in_sizes; (void)n_in; (void)out_size; (void)ws_size;
    const float* h      = (const float*)d_in[0];
    const float* e      = (const float*)d_in[1];
    const float* ln_h_g = (const float*)d_in[2];
    const float* ln_h_b = (const float*)d_in[3];
    const float* ln_e_g = (const float*)d_in[4];
    const float* ln_e_b = (const float*)d_in[5];
    const float* W_E    = (const float*)d_in[6];
    const float* b_E    = (const float*)d_in[7];
    const float* W_QKV  = (const float*)d_in[8];
    const float* b_QKV  = (const float*)d_in[9];
    const float* W_G    = (const float*)d_in[10];
    const float* b_G    = (const float*)d_in[11];
    const float* W_Oh   = (const float*)d_in[12];
    const float* b_Oh   = (const float*)d_in[13];
    const float* fln_h_g= (const float*)d_in[14];
    const float* fln_h_b= (const float*)d_in[15];
    const float* W_h1   = (const float*)d_in[16];
    const float* b_h1   = (const float*)d_in[17];
    const float* W_h2   = (const float*)d_in[18];
    const float* b_h2   = (const float*)d_in[19];
    const float* W_Oe   = (const float*)d_in[20];
    const float* b_Oe   = (const float*)d_in[21];
    const float* fln_e_g= (const float*)d_in[22];
    const float* fln_e_b= (const float*)d_in[23];
    const float* W_e1   = (const float*)d_in[24];
    const float* b_e1   = (const float*)d_in[25];
    const float* W_e2   = (const float*)d_in[26];
    const float* b_e2   = (const float*)d_in[27];

    char* ws = (char*)d_ws;
    size_t off = 0;
    auto alloc = [&](size_t bytes) -> char* {
        char* p = ws + off;
        off = (off + bytes + 255) & ~(size_t)255;
        return p;
    };
    _Float16* hln16 = (_Float16*)alloc(2048ull * 1024 * 2);
    float*    QKV32 = (float*)   alloc(2048ull * 3072 * 4);
    _Float16* Qf    = (_Float16*)alloc(2097152ull * 2);
    _Float16* KTp   = (_Float16*)alloc(2097152ull * 2);
    _Float16* Vp    = (_Float16*)alloc(2097152ull * 2);
    float*    ET    = (float*)   alloc(16777216ull * 4);
    float*    GT    = (float*)   alloc(16777216ull * 4);
    float*    Hh    = (float*)   alloc(16777216ull * 4);
    _Float16* At16  = (_Float16*)alloc(16777216ull * 2);
    _Float16* Vatt  = (_Float16*)alloc(2048ull * 1024 * 2);
    float*    hout1 = (float*)   alloc(2048ull * 1024 * 4);
    _Float16* h216  = (_Float16*)alloc(2048ull * 1024 * 2);
    _Float16* hid16 = (_Float16*)alloc(2048ull * 2048 * 2);
    _Float16* WQKVp = (_Float16*)alloc(1024ull * 3072 * 2);
    _Float16* WOhp  = (_Float16*)alloc(1024ull * 1024 * 2);
    _Float16* Wh1p  = (_Float16*)alloc(1024ull * 2048 * 2);
    _Float16* Wh2p  = (_Float16*)alloc(2048ull * 1024 * 2);
    _Float16* We1p  = (_Float16*)alloc(64ull * 128 * 2);
    _Float16* We2p  = (_Float16*)alloc(128ull * 64 * 2);

    float* outh = (float*)d_out;
    float* oute = outh + 2048ull * 1024;

    // Pack weights into WMMA B-fragment order (cheap; redone every call).
    pack_b<<<(1024 * 3072) / 256, 256, 0, stream>>>(W_QKV, WQKVp, 1024, 3072);
    pack_b<<<(1024 * 1024) / 256, 256, 0, stream>>>(W_Oh, WOhp, 1024, 1024);
    pack_b<<<(1024 * 2048) / 256, 256, 0, stream>>>(W_h1, Wh1p, 1024, 2048);
    pack_b<<<(2048 * 1024) / 256, 256, 0, stream>>>(W_h2, Wh2p, 2048, 1024);
    pack_b<<<(64 * 128) / 256, 256, 0, stream>>>(W_e1, We1p, 64, 128);
    pack_b<<<(128 * 64) / 256, 256, 0, stream>>>(W_e2, We2p, 128, 64);

    // h path up to attention
    ln_rows<<<2048, 256, 0, stream>>>(h, ln_h_g, ln_h_b, hln16, 1024);
    gemm_wmma<false, false, true, false><<<3072, 256, 0, stream>>>(
        hln16, WQKVp, b_QKV, nullptr, QKV32, nullptr, 2048, 3072, 1024);
    prep_qkv<<<dim3(384, 64), 256, 0, stream>>>(QKV32, Qf, KTp, Vp);

    // e path: E and gates
    eg_kernel<<<16384, 256, 0, stream>>>(e, ln_e_g, ln_e_b, W_E, b_E, W_G, b_G, ET, GT);

    // attention
    scores_kernel<<<dim3(128, 64), 256, 0, stream>>>(Qf, KTp, ET, Hh);
    softmax_kernel<<<4096, 256, 0, stream>>>(Hh, GT, At16);
    attnv_kernel<<<dim3(16, 64), 256, 0, stream>>>(At16, Vp, Vatt);

    // h output projection + FFN
    gemm_wmma<true, false, true, false><<<1024, 256, 0, stream>>>(
        Vatt, WOhp, b_Oh, h, hout1, nullptr, 2048, 1024, 1024);
    ln_rows<<<2048, 256, 0, stream>>>(hout1, fln_h_g, fln_h_b, h216, 1024);
    gemm_wmma<false, true, false, true><<<2048, 256, 0, stream>>>(
        h216, Wh1p, b_h1, nullptr, nullptr, hid16, 2048, 2048, 1024);
    gemm_wmma<true, false, true, false><<<1024, 256, 0, stream>>>(
        hid16, Wh2p, b_h2, hout1, outh, nullptr, 2048, 1024, 2048);

    // e output projection + fused FFN (in place in d_out)
    eout1_kernel<<<262144, 256, 0, stream>>>(Hh, e, W_Oe, b_Oe, oute);
    effn_kernel<<<65536, 128, 0, stream>>>(oute, fln_e_g, fln_e_b, We1p, b_e1, We2p, b_e2);
}
